// TSLM_78099685310915
// MI455X (gfx1250) — matrix operand order repaired and checked by reading, not compiled, not run on hardware
//
#include <hip/hip_runtime.h>

// ---------------------------------------------------------------------------
// Transformer forward for MI455X (gfx1250, wave32, WMMA).
// - All matmuls: v_wmma_f32_16x16x32_bf16 (bf16 in, fp32 accumulate).
// - GEMM: 128x128 block tiles, LDS ping-pong software pipeline.
// - Q/K/V and FFN-mid tensors carried in bf16 memory (halves traffic).
// - Attention: fused flash-style; K tile staged via async global->LDS DMA.
// ---------------------------------------------------------------------------

#define DEV __device__ __forceinline__

constexpr int Vv = 32000;
constexpr int Dm = 1024;
constexpr int Hh = 16;
constexpr int Ll = 6;
constexpr int Dff = 4096;
constexpr int Bb = 2;
constexpr int Ss = 2048;
constexpr int Dk = 64;
constexpr int Mrows = Bb * Ss;  // 4096

// epilogue mode bits
#define MODE_RELU 1
#define MODE_HEADS 2
#define MODE_BIAS 4
#define MODE_RES 8

typedef __attribute__((ext_vector_type(16))) __bf16 bf16x16;
typedef __attribute__((ext_vector_type(2))) __bf16 bf16x2;
typedef __attribute__((ext_vector_type(2))) float f32x2;
typedef __attribute__((ext_vector_type(8))) float f32x8;

struct Frag { unsigned int r[8]; };  // 16 bf16 values (A or B operand, per lane)

// fp32x2 -> packed bf16x2 (packed cvt path on gfx1250)
DEV unsigned int pack2(float a, float b) {
  f32x2 v = {a, b};
  bf16x2 r = __builtin_convertvector(v, bf16x2);
  return __builtin_bit_cast(unsigned int, r);
}
DEV unsigned short f2bf(float f) {
  __bf16 h = (__bf16)f;
  return __builtin_bit_cast(unsigned short, h);
}
DEV void packstore(unsigned short* dst, const float4& a, const float4& b) {
  unsigned int t[4];
  t[0] = pack2(a.x, a.y);
  t[1] = pack2(a.z, a.w);
  t[2] = pack2(b.x, b.y);
  t[3] = pack2(b.z, b.w);
  *(uint4*)dst = *(uint4*)t;
}
// A-operand fragment (16x32 bf16): lane holds row (lane&15), K runs at
// half*8 and 16+half*8 within the 32-wide K slice.
DEV Frag ldsA(const unsigned short* rowp, int half) {
  Frag f;
  *(uint4*)&f.r[0] = *(const uint4*)(rowp + half * 8);
  *(uint4*)&f.r[4] = *(const uint4*)(rowp + 16 + half * 8);
  return f;
}
// B-operand fragment (32x16 bf16): lane holds col (lane&15), K runs at
// half*8 and 16+half*8 within the 32-wide K slice.
DEV Frag ldsB(const unsigned short* rowp, int half) {
  Frag f;
  *(uint4*)&f.r[0] = *(const uint4*)(rowp + half * 8);
  *(uint4*)&f.r[4] = *(const uint4*)(rowp + 16 + half * 8);
  return f;
}
DEV f32x8 wmma_bf16(const Frag& a, const Frag& b, f32x8 c) {
  return __builtin_amdgcn_wmma_f32_16x16x32_bf16(
      false, __builtin_bit_cast(bf16x16, a),
      false, __builtin_bit_cast(bf16x16, b),
      (short)0, c, false, false);
}
DEV f32x8 fzero() {
  f32x8 z;
#pragma unroll
  for (int i = 0; i < 8; ++i) z[i] = 0.f;
  return z;
}
// Async global -> LDS DMA (CDNA5, tracked by ASYNCcnt). Low 32 bits of a
// generic pointer to __shared__ give the LDS byte offset (flat aperture
// truncation, ISA 10.2).
DEV void async_load_b128(unsigned lds_off, const void* gaddr) {
  asm volatile("global_load_async_to_lds_b128 %0, %1, off"
               :
               : "v"(lds_off), "v"(gaddr)
               : "memory");
}
DEV void wait_async0() { asm volatile("s_wait_asynccnt 0" ::: "memory"); }

// ---------------------------------------------------------------------------
// Embedding + sinusoidal positional encoding: h = emb[x]*sqrt(D) + PE
// ---------------------------------------------------------------------------
__global__ __launch_bounds__(256) void embed_k(const int* __restrict__ x,
                                               const float* __restrict__ emb,
                                               float* __restrict__ h) {
  const int s = blockIdx.x, b = blockIdx.y;
  const int tok = x[b * Ss + s];
  const int d0 = threadIdx.x * 4;
  const float* ep = emb + (size_t)tok * Dm + d0;
  float* hp = h + ((size_t)(b * Ss + s)) * Dm + d0;
#pragma unroll
  for (int i = 0; i < 4; ++i) {
    const int d = d0 + i;
    const int pi = d >> 1;
    const float freq = __expf((float)(2 * pi) * (-9.210340371976184f / (float)Dm));
    const float ang = (float)s * freq;
    const float pe = (d & 1) ? __cosf(ang) : __sinf(ang);
    hp[i] = ep[i] * 32.0f + pe;  // sqrt(1024) = 32
  }
}

// ---------------------------------------------------------------------------
// GEMM: out[M,N] = A[M,K] * W[N,K]^T (+bias[N]) (+residual[M,N]) (relu)
// Block tile 128x128, 8 waves (2 in M x 4 in N), 8 WMMAs/wave per 32-wide
// k-step. Two-deep LDS ping-pong pipeline. ABF: A operand is bf16 in memory
// (raw uint4 staging, no conversion). OBF: store output as bf16.
// ---------------------------------------------------------------------------
template <int MODE, bool ABF, bool OBF>
__global__ __launch_bounds__(256) void gemm_wmma(
    const void* __restrict__ Av, const float* __restrict__ W,
    const float* __restrict__ bias, const float* __restrict__ residual,
    void* __restrict__ outv, int M, int N, int K) {
  __shared__ __align__(16) unsigned short As[2][128 * 32];
  __shared__ __align__(16) unsigned short Bs[2][128 * 32];
  const int tid = threadIdx.x, lane = tid & 31, wid = tid >> 5;
  const int half = lane >> 4, mr = lane & 15;
  const int wm = wid >> 2, wn = wid & 3;  // 2 x 4 wave grid
  const int m0 = blockIdx.y * 128, n0 = blockIdx.x * 128;
  const int lr = tid >> 2, lc = (tid & 3) * 8;  // coop tile load: 2x(64x32)
  f32x8 acc[8];
#pragma unroll
  for (int i = 0; i < 8; ++i) acc[i] = fzero();

  const float* Af = (const float*)Av;
  const unsigned short* Ah = (const unsigned short*)Av;
  const float* arow0f = Af + (size_t)(m0 + lr) * K + lc;
  const float* arow1f = Af + (size_t)(m0 + lr + 64) * K + lc;
  const unsigned short* arow0h = Ah + (size_t)(m0 + lr) * K + lc;
  const unsigned short* arow1h = Ah + (size_t)(m0 + lr + 64) * K + lc;
  const float* wrow0 = W + (size_t)(n0 + lr) * K + lc;
  const float* wrow1 = W + (size_t)(n0 + lr + 64) * K + lc;

  float4 paf[4], pbf[4];
  uint4 pah[2];
  auto prefetch = [&](int ko) {
    if constexpr (ABF) {
      pah[0] = *(const uint4*)(arow0h + ko);
      pah[1] = *(const uint4*)(arow1h + ko);
    } else {
      paf[0] = *(const float4*)(arow0f + ko);
      paf[1] = *(const float4*)(arow0f + ko + 4);
      paf[2] = *(const float4*)(arow1f + ko);
      paf[3] = *(const float4*)(arow1f + ko + 4);
    }
    pbf[0] = *(const float4*)(wrow0 + ko);
    pbf[1] = *(const float4*)(wrow0 + ko + 4);
    pbf[2] = *(const float4*)(wrow1 + ko);
    pbf[3] = *(const float4*)(wrow1 + ko + 4);
  };
  prefetch(0);

  int p = 0;
  for (int k0 = 0; k0 < K; k0 += 32) {
    // stage prefetched k-step into LDS buffer p (bf16)
    if constexpr (ABF) {
      *(uint4*)&As[p][lr * 32 + lc] = pah[0];
      *(uint4*)&As[p][(lr + 64) * 32 + lc] = pah[1];
    } else {
      packstore(&As[p][lr * 32 + lc], paf[0], paf[1]);
      packstore(&As[p][(lr + 64) * 32 + lc], paf[2], paf[3]);
    }
    packstore(&Bs[p][lr * 32 + lc], pbf[0], pbf[1]);
    packstore(&Bs[p][(lr + 64) * 32 + lc], pbf[2], pbf[3]);
    __syncthreads();
    if (k0 + 32 < K) prefetch(k0 + 32);  // overlap with WMMA below
    Frag fa[4], fb[2];
#pragma unroll
    for (int i = 0; i < 4; ++i)
      fa[i] = ldsA(&As[p][(wm * 64 + i * 16 + mr) * 32], half);
#pragma unroll
    for (int j = 0; j < 2; ++j)
      fb[j] = ldsB(&Bs[p][(wn * 32 + j * 16 + mr) * 32], half);
#pragma unroll
    for (int i = 0; i < 4; ++i)
#pragma unroll
      for (int j = 0; j < 2; ++j)
        acc[i * 2 + j] = wmma_bf16(fa[i], fb[j], acc[i * 2 + j]);
    p ^= 1;
  }

  float* outf = (float*)outv;
  unsigned short* outh = (unsigned short*)outv;
#pragma unroll
  for (int i = 0; i < 4; ++i) {
#pragma unroll
    for (int j2 = 0; j2 < 2; ++j2) {
      const f32x8 a = acc[i * 2 + j2];
      const int col = n0 + wn * 32 + j2 * 16 + mr;
      const float bv = (MODE & MODE_BIAS) ? bias[col] : 0.f;
#pragma unroll
      for (int j = 0; j < 8; ++j) {
        const int row = m0 + wm * 64 + i * 16 + half * 8 + j;
        float v = a[j] + bv;
        if (MODE & MODE_RES) v += residual[(size_t)row * N + col];
        if (MODE & MODE_RELU) v = fmaxf(v, 0.f);
        if constexpr (MODE & MODE_HEADS) {
          const int bb = row / Ss, ss = row % Ss;
          const int hh = col / Dk, dd = col % Dk;
          outh[(((size_t)bb * Hh + hh) * Ss + ss) * Dk + dd] = f2bf(v);
        } else if constexpr (OBF) {
          outh[(size_t)row * N + col] = f2bf(v);
        } else {
          outf[(size_t)row * N + col] = v;
        }
      }
    }
  }
}

// ---------------------------------------------------------------------------
// Fused causal attention (flash-style, online softmax).
// Q,K,V: [B,H,S,DK] bf16. Block = 128 query rows of one (b,h); 8 waves, each
// wave owns 16 query rows. K tile DMA'd to LDS with async-to-LDS (no VGPR
// round trip, ASYNCcnt); V loaded and transposed into LDS; P goes through a
// wave-private LDS scratch for C-layout -> A-layout. Output -> [B,S,D] fp32.
// ---------------------------------------------------------------------------
__global__ __launch_bounds__(256) void attn_wmma(
    const unsigned short* __restrict__ Qb, const unsigned short* __restrict__ Kb,
    const unsigned short* __restrict__ Vb, float* __restrict__ ctx) {
  __shared__ __align__(16) unsigned short Ks[64 * 64];      // [k][dk]
  __shared__ __align__(16) unsigned short Vt[64 * 64];      // [dk][k]
  __shared__ __align__(16) unsigned short Ps[8 * 16 * 64];  // per-wave P scratch
  const int tid = threadIdx.x, lane = tid & 31, wid = tid >> 5;
  const int half = lane >> 4, mr = lane & 15;
  const int bh = blockIdx.y;
  const size_t base = (size_t)bh * Ss * Dk;
  const int q0 = blockIdx.x * 128 + wid * 16;

  // Q fragments: direct 16B bf16 loads (per-lane runs are 16B aligned)
  Frag qf[2];
  {
    const unsigned short* qrow = Qb + base + (size_t)(q0 + mr) * Dk;
#pragma unroll
    for (int ks = 0; ks < 2; ++ks) {
      *(uint4*)&qf[ks].r[0] = *(const uint4*)(qrow + ks * 32 + half * 8);
      *(uint4*)&qf[ks].r[4] = *(const uint4*)(qrow + ks * 32 + 16 + half * 8);
    }
  }
  float mi[8], li[8];
#pragma unroll
  for (int j = 0; j < 8; ++j) { mi[j] = -1e30f; li[j] = 0.f; }
  f32x8 o[4];
#pragma unroll
  for (int dt = 0; dt < 4; ++dt) o[dt] = fzero();
  unsigned short* Pw = &Ps[wid * (16 * 64)];

  const int lrow = tid >> 2, lcb = (tid & 3) * 16;  // cooperative 64x64 load
  const int ktEnd = blockIdx.x * 2 + 2;             // causal bound for block
  for (int kt = 0; kt < ktEnd; ++kt) {
    {
      // K tile: async DMA straight into LDS (row-major [k][dk] is already
      // the B-operand layout for QK^T)
      const unsigned short* kp = Kb + base + (size_t)(kt * 64 + lrow) * Dk + lcb;
      const unsigned lds_off = (unsigned)(uintptr_t)&Ks[lrow * 64 + lcb];
      async_load_b128(lds_off, kp);
      async_load_b128(lds_off + 16, kp + 8);
      // V tile: load bf16, transpose-scatter into LDS [dk][k]
      const unsigned short* vp = Vb + base + (size_t)(kt * 64 + lrow) * Dk + lcb;
      const uint4 v0 = *(const uint4*)vp;
      const uint4 v1 = *(const uint4*)(vp + 8);
      const unsigned w[8] = {v0.x, v0.y, v0.z, v0.w, v1.x, v1.y, v1.z, v1.w};
#pragma unroll
      for (int i = 0; i < 16; ++i)
        Vt[(lcb + i) * 64 + lrow] =
            (unsigned short)(w[i >> 1] >> ((i & 1) * 16));
    }
    wait_async0();
    __syncthreads();

    // S = Q K^T for this wave's 16 rows x 64 keys (4 tiles, 2 k-steps)
    f32x8 s[4];
#pragma unroll
    for (int nt = 0; nt < 4; ++nt) {
      s[nt] = fzero();
#pragma unroll
      for (int ks = 0; ks < 2; ++ks) {
        Frag kb = ldsB(&Ks[(nt * 16 + mr) * 64 + ks * 32], half);
        s[nt] = wmma_bf16(qf[ks], kb, s[nt]);
      }
    }
    // scale + mask + online softmax, row-wise
#pragma unroll
    for (int j = 0; j < 8; ++j) {
      const int qr = q0 + half * 8 + j;
      float mx = -1e30f;
#pragma unroll
      for (int nt = 0; nt < 4; ++nt) {
        const int kc = kt * 64 + nt * 16 + mr;
        float v = s[nt][j] * 0.125f;  // 1/sqrt(64)
        if (kc > qr) v = -1e9f;       // matches reference causal fill
        s[nt][j] = v;
        mx = fmaxf(mx, v);
      }
      mx = fmaxf(mx, __shfl_xor(mx, 1, 32));
      mx = fmaxf(mx, __shfl_xor(mx, 2, 32));
      mx = fmaxf(mx, __shfl_xor(mx, 4, 32));
      mx = fmaxf(mx, __shfl_xor(mx, 8, 32));
      const float mn = fmaxf(mi[j], mx);
      const float corr = __expf(mi[j] - mn);
      float ls = 0.f;
#pragma unroll
      for (int nt = 0; nt < 4; ++nt) {
        const float pv = __expf(s[nt][j] - mn);
        s[nt][j] = pv;
        ls += pv;
      }
      ls += __shfl_xor(ls, 1, 32);
      ls += __shfl_xor(ls, 2, 32);
      ls += __shfl_xor(ls, 4, 32);
      ls += __shfl_xor(ls, 8, 32);
      li[j] = li[j] * corr + ls;
      mi[j] = mn;
#pragma unroll
      for (int dt = 0; dt < 4; ++dt) o[dt][j] *= corr;
#pragma unroll
      for (int nt = 0; nt < 4; ++nt)
        Pw[(half * 8 + j) * 64 + nt * 16 + mr] = f2bf(s[nt][j]);
    }
    // O += P V  (P re-read from wave-private LDS in A-layout)
#pragma unroll
    for (int ks = 0; ks < 2; ++ks) {
      Frag pf = ldsA(&Pw[mr * 64 + ks * 32], half);
#pragma unroll
      for (int dt = 0; dt < 4; ++dt) {
        Frag vb = ldsB(&Vt[(dt * 16 + mr) * 64 + ks * 32], half);
        o[dt] = wmma_bf16(pf, vb, o[dt]);
      }
    }
    __syncthreads();
  }
  // normalize and scatter to [B,S,D]
  const int b = bh / Hh, hd = bh % Hh;
#pragma unroll
  for (int dt = 0; dt < 4; ++dt) {
#pragma unroll
    for (int j = 0; j < 8; ++j) {
      const int srow = q0 + half * 8 + j;
      const float v = o[dt][j] / li[j];
      ctx[((size_t)(b * Ss + srow)) * Dm + hd * Dk + dt * 16 + mr] = v;
    }
  }
}

// ---------------------------------------------------------------------------
// LayerNorm over D=1024, one block (256 threads) per row.
// ---------------------------------------------------------------------------
__global__ __launch_bounds__(256) void layernorm_k(const float* __restrict__ in,
                                                   const float* __restrict__ g,
                                                   const float* __restrict__ bta,
                                                   float* __restrict__ out) {
  const int row = blockIdx.x;
  const int tid = threadIdx.x, lane = tid & 31, wid = tid >> 5;
  const int d0 = tid * 4;
  const float4 v = *(const float4*)(in + (size_t)row * Dm + d0);
  float s = v.x + v.y + v.z + v.w;
  float s2 = v.x * v.x + v.y * v.y + v.z * v.z + v.w * v.w;
#pragma unroll
  for (int m = 16; m >= 1; m >>= 1) {
    s += __shfl_xor(s, m, 32);
    s2 += __shfl_xor(s2, m, 32);
  }
  __shared__ float r1[8], r2[8];
  if (lane == 0) { r1[wid] = s; r2[wid] = s2; }
  __syncthreads();
  s = 0.f;
  s2 = 0.f;
#pragma unroll
  for (int i = 0; i < 8; ++i) { s += r1[i]; s2 += r2[i]; }
  const float mu = s * (1.f / Dm);
  const float inv = rsqrtf(s2 * (1.f / Dm) - mu * mu + 1e-5f);
  const float4 go = *(const float4*)(g + d0);
  const float4 bo = *(const float4*)(bta + d0);
  float4 r;
  r.x = (v.x - mu) * inv * go.x + bo.x;
  r.y = (v.y - mu) * inv * go.y + bo.y;
  r.z = (v.z - mu) * inv * go.z + bo.z;
  r.w = (v.w - mu) * inv * go.w + bo.w;
  *(float4*)(out + (size_t)row * Dm + d0) = r;
}

// ---------------------------------------------------------------------------
// Host-side orchestration (all on `stream`, graph-capture safe).
// ---------------------------------------------------------------------------
extern "C" void kernel_launch(void* const* d_in, const int* in_sizes, int n_in,
                              void* d_out, int out_size, void* d_ws,
                              size_t ws_size, hipStream_t stream) {
  (void)in_sizes; (void)n_in; (void)out_size; (void)ws_size;
  const int* x = (const int*)d_in[0];
  const float* emb = (const float*)d_in[1];
  const float* Wq = (const float*)d_in[2];
  const float* Wk = (const float*)d_in[3];
  const float* Wv = (const float*)d_in[4];
  const float* Wo = (const float*)d_in[5];
  const float* f1w = (const float*)d_in[6];
  const float* f1b = (const float*)d_in[7];
  const float* f2w = (const float*)d_in[8];
  const float* f2b = (const float*)d_in[9];
  const float* g1 = (const float*)d_in[10];
  const float* b1 = (const float*)d_in[11];
  const float* g2 = (const float*)d_in[12];
  const float* b2 = (const float*)d_in[13];
  const float* fng = (const float*)d_in[14];
  const float* fnb = (const float*)d_in[15];
  const float* ow = (const float*)d_in[16];
  const float* ob = (const float*)d_in[17];

  char* ws = (char*)d_ws;
  const size_t MB = (size_t)1 << 20;
  float* h = (float*)(ws + 0 * MB);                    // [4096,1024] fp32 16MB
  unsigned short* t0 = (unsigned short*)(ws + 16 * MB);  // [4096,4096] bf16 32MB
  unsigned short* qb = (unsigned short*)(ws + 48 * MB);  // [B,H,S,DK] bf16 8MB
  unsigned short* kb = (unsigned short*)(ws + 56 * MB);
  unsigned short* vb = (unsigned short*)(ws + 64 * MB);
  float* ctx = (float*)(ws + 72 * MB);                 // [B,S,D] fp32 16MB
  float* h1 = (float*)(ws + 88 * MB);
  float* pre = (float*)(ws + 104 * MB);

  const dim3 blk(256);
  embed_k<<<dim3(Ss, Bb), blk, 0, stream>>>(x, emb, h);

  const dim3 gD(Dm / 128, Mrows / 128);   // D-wide GEMMs
  const dim3 gF(Dff / 128, Mrows / 128);  // FFN-up GEMM
  const dim3 gV(Vv / 128, Mrows / 128);   // vocab GEMM

  for (int l = 0; l < Ll; ++l) {
    const float* wq = Wq + (size_t)l * Dm * Dm;
    const float* wk = Wk + (size_t)l * Dm * Dm;
    const float* wv = Wv + (size_t)l * Dm * Dm;
    const float* wo = Wo + (size_t)l * Dm * Dm;
    gemm_wmma<MODE_HEADS, false, true><<<gD, blk, 0, stream>>>(
        h, wq, nullptr, nullptr, qb, Mrows, Dm, Dm);
    gemm_wmma<MODE_HEADS, false, true><<<gD, blk, 0, stream>>>(
        h, wk, nullptr, nullptr, kb, Mrows, Dm, Dm);
    gemm_wmma<MODE_HEADS, false, true><<<gD, blk, 0, stream>>>(
        h, wv, nullptr, nullptr, vb, Mrows, Dm, Dm);
    attn_wmma<<<dim3(Ss / 128, Bb * Hh), blk, 0, stream>>>(qb, kb, vb, ctx);
    gemm_wmma<MODE_RES, false, false><<<gD, blk, 0, stream>>>(
        ctx, wo, nullptr, h, pre, Mrows, Dm, Dm);
    layernorm_k<<<Mrows, blk, 0, stream>>>(pre, g1 + l * Dm, b1 + l * Dm, h1);
    gemm_wmma<MODE_BIAS | MODE_RELU, false, true><<<gF, blk, 0, stream>>>(
        h1, f1w + (size_t)l * Dff * Dm, f1b + (size_t)l * Dff, nullptr, t0,
        Mrows, Dff, Dm);
    gemm_wmma<MODE_BIAS | MODE_RES, true, false><<<gD, blk, 0, stream>>>(
        t0, f2w + (size_t)l * Dm * Dff, f2b + (size_t)l * Dm, h1, pre, Mrows,
        Dm, Dff);
    layernorm_k<<<Mrows, blk, 0, stream>>>(pre, g2 + l * Dm, b2 + l * Dm, h);
  }
  layernorm_k<<<Mrows, blk, 0, stream>>>(h, fng, fnb, h1);
  gemm_wmma<MODE_BIAS, false, false><<<gV, blk, 0, stream>>>(
      h1, ow, ob, nullptr, (float*)d_out, Mrows, Vv, Dm);
}
